// HNHN_46574625357936
// MI455X (gfx1250) — matrix operand order repaired and checked by reading, not compiled, not run on hardware
//
#include <hip/hip_runtime.h>
#include <cstdint>
#include <cstddef>

typedef __attribute__((ext_vector_type(16))) __bf16 v16bf;
typedef __attribute__((ext_vector_type(8)))  float  v8f;

#define D_DIM 128

__device__ __forceinline__ unsigned short f32_to_bf16(float f) {
    unsigned int u = __builtin_bit_cast(unsigned int, f);
    u += 0x7FFFu + ((u >> 16) & 1u);   // round-to-nearest-even
    return (unsigned short)(u >> 16);
}

// ---------------------------------------------------------------- W_c = Wve @ W1  (f32, tiny)
__global__ __launch_bounds__(256) void fuse_weights(const float* __restrict__ Wve,
                                                    const float* __restrict__ W1,
                                                    float* __restrict__ Wc) {
    const int idx = blockIdx.x * blockDim.x + threadIdx.x;   // 0..16383
    const int r = idx >> 7, c = idx & 127;
    float acc = 0.f;
#pragma unroll 8
    for (int k = 0; k < 128; ++k) acc += Wve[r * 128 + k] * W1[k * 128 + c];
    Wc[idx] = acc;
}

// ---------------------------------------------------------------- b_c = Wve @ b1 + bve
__global__ __launch_bounds__(128) void fuse_bias(const float* __restrict__ Wve,
                                                 const float* __restrict__ b1,
                                                 const float* __restrict__ bve,
                                                 float* __restrict__ bc) {
    const int j = threadIdx.x;
    float acc = bve[j];
#pragma unroll 8
    for (int k = 0; k < 128; ++k) acc += Wve[j * 128 + k] * b1[k];
    bc[j] = acc;
}

// ---------------------------------------------------------------- convert f32 -> bf16 (x4 vectorized)
__global__ __launch_bounds__(256) void cvt4_f32_bf16(const float* __restrict__ in,
                                                     unsigned short* __restrict__ out,
                                                     long long n4) {
    long long i = (long long)blockIdx.x * blockDim.x + threadIdx.x;
    if (i < n4) {
        float4 v = ((const float4*)in)[i];
        ushort4 o;
        o.x = f32_to_bf16(v.x); o.y = f32_to_bf16(v.y);
        o.z = f32_to_bf16(v.z); o.w = f32_to_bf16(v.w);
        ((ushort4*)out)[i] = o;
    }
}

// ---------------------------------------------------------------- zero f32 (x4 vectorized)
__global__ __launch_bounds__(256) void zero4_f32(float* __restrict__ p, long long n4) {
    long long i = (long long)blockIdx.x * blockDim.x + threadIdx.x;
    const float4 z = {0.f, 0.f, 0.f, 0.f};
    if (i < n4) ((float4*)p)[i] = z;
}

// ---------------------------------------------------------------- WMMA GEMM
// out[M,128] (f32) = A_bf16[M,128] @ W^T + bias ; W is [128,128] row-major bf16.
// 256 threads = 8 waves; each wave computes a 16-row strip across all 8
// column tiles with v_wmma_f32_16x16x32_bf16, W staged once per block in LDS.
__global__ __launch_bounds__(256) void gemm_wmma_bf16(
        const unsigned short* __restrict__ A,
        const unsigned short* __restrict__ W,
        const float* __restrict__ bias,
        float* __restrict__ outF,
        int M) {
    __shared__ unsigned short sW[128][136];   // +8 shorts pad: bank spread, 16B aligned

    for (int idx = threadIdx.x; idx < 128 * 128; idx += 256) {
        int r = idx >> 7, c = idx & 127;
        sW[r][c] = W[idx];
    }
    __syncthreads();

    const int wave   = threadIdx.x >> 5;
    const int lane   = threadIdx.x & 31;
    const int tile_m = blockIdx.x * 8 + wave;
    if (tile_m * 16 >= M) return;           // after barrier: safe

    const int ml = lane & 15;               // row-in-tile (A) / col-in-tile (B,C)
    const int hi = lane >> 4;               // lane-half selector

    const unsigned short* Arow = A + (size_t)(tile_m * 16 + ml) * D_DIM;

    v8f acc[8];
    const v8f vzero = {0.f, 0.f, 0.f, 0.f, 0.f, 0.f, 0.f, 0.f};
#pragma unroll
    for (int n = 0; n < 8; ++n) acc[n] = vzero;

#pragma unroll
    for (int ks = 0; ks < 4; ++ks) {
        const int kb = ks * 32;
        // A fragment: 16x32 bf16 per ISA layout
        union { uint4 q[2]; v16bf v; } au;
        au.q[0] = *(const uint4*)(Arow + kb + hi * 8);
        au.q[1] = *(const uint4*)(Arow + kb + 16 + hi * 8);
#pragma unroll
        for (int n = 0; n < 8; ++n) {
            // B fragment: B[k][n] = W[n][k]; lane covers K = kb + hi*8 .. +7 and kb + 16 + hi*8 .. +7
            const unsigned short* bp = &sW[n * 16 + ml][kb];
            union { uint4 q[2]; v16bf v; } bu;
            bu.q[0] = *(const uint4*)(bp + hi * 8);
            bu.q[1] = *(const uint4*)(bp + 16 + hi * 8);
            acc[n] = __builtin_amdgcn_wmma_f32_16x16x32_bf16(
                false, au.v, false, bu.v, (short)0, acc[n], false, false);
        }
    }

    // store: VGPR r -> (M = r + 8*hi, N = ml)
#pragma unroll
    for (int n = 0; n < 8; ++n) {
        const int col = n * 16 + ml;
        const float bv = bias[col];
#pragma unroll
        for (int r = 0; r < 8; ++r) {
            const int row = tile_m * 16 + r + (hi << 3);
            outF[(size_t)row * D_DIM + col] = acc[n][r] + bv;
        }
    }
}

// ---------------------------------------------------------------- scatter-add
// One wave per incidence i; indices + weight forced scalar (wave-uniform).
__global__ __launch_bounds__(256) void scatter_add(
        const float* __restrict__ src,
        const int* __restrict__ src_idx,
        const int* __restrict__ dst_idx,
        const float* __restrict__ wnum,
        const float* __restrict__ wden,
        float* __restrict__ out,
        int NI) {
    const int i = blockIdx.x * 8 + (threadIdx.x >> 5);
    if (i >= NI) return;
    const int lane = threadIdx.x & 31;
    const int s = __builtin_amdgcn_readfirstlane(src_idx[i]);
    const int d = __builtin_amdgcn_readfirstlane(dst_idx[i]);
    const float w = wnum[s] / wden[d];     // uniform: scalar loads + one divide
    const float4 v = *((const float4*)(src + (size_t)s * D_DIM) + lane);
    float* op = out + (size_t)d * D_DIM + lane * 4;
    atomicAdd(op + 0, v.x * w);
    atomicAdd(op + 1, v.y * w);
    atomicAdd(op + 2, v.z * w);
    atomicAdd(op + 3, v.w * w);
}

// ================================================================ host
extern "C" void kernel_launch(void* const* d_in, const int* in_sizes, int n_in,
                              void* d_out, int out_size, void* d_ws, size_t ws_size,
                              hipStream_t stream) {
    const float* vfeat        = (const float*)d_in[0];
    // d_in[1] (efeat) unused by reference
    const float* v_reg_weight = (const float*)d_in[2];
    const float* v_reg_sum    = (const float*)d_in[3];
    const float* e_reg_weight = (const float*)d_in[4];
    const float* e_reg_sum    = (const float*)d_in[5];
    const int*   node_idx     = (const int*)d_in[6];
    const int*   hedge_idx    = (const int*)d_in[7];
    const float* W1  = (const float*)d_in[8];
    const float* b1  = (const float*)d_in[9];
    const float* Wve = (const float*)d_in[10];
    const float* bve = (const float*)d_in[11];
    const float* Wev = (const float*)d_in[12];
    const float* bev = (const float*)d_in[13];

    const int NV = in_sizes[0] / D_DIM;
    const int NE = in_sizes[1] / D_DIM;
    const int NI = in_sizes[6];
    const long long nvElems = (long long)NV * D_DIM;
    const long long neElems = (long long)NE * D_DIM;

    // ---- workspace layout (bytes, 256-aligned); NE buffers overlay dead vfeat_bf16
    char* ws = (char*)d_ws;
    size_t off = 0;
    auto take = [&](size_t bytes) { size_t r = off; off = (off + bytes + 255) & ~(size_t)255; return r; };
    auto al   = [](size_t b) { return (b + 255) & ~(size_t)255; };

    const size_t oWc   = take(128 * 128 * 4);          // fused weights f32
    const size_t oBc   = take(128 * 4);                // fused bias f32
    const size_t oWcb  = take(128 * 128 * 2);          // fused weights bf16
    const size_t oWevb = take(128 * 128 * 2);          // Wev bf16
    const size_t oWhn  = take((size_t)nvElems * 4);    // Wh_n f32
    const size_t uBase = off;                          // overlay region start
    const size_t oVfB  = uBase;                        // vfeat bf16 (dead after fused GEMM)
    const size_t oFe   = uBase;                        // feat_e f32
    const size_t oFeB  = oFe  + al((size_t)neElems * 4);   // feat_e bf16
    const size_t oWhe  = oFeB + al((size_t)neElems * 2);   // Wh_e f32
    size_t uEndA = oVfB + al((size_t)nvElems * 2);
    size_t uEndB = oWhe + al((size_t)neElems * 4);
    off = (uEndA > uEndB) ? uEndA : uEndB;
    (void)ws_size; (void)n_in; (void)out_size; (void)off;

    float*          Wc   = (float*)(ws + oWc);
    float*          bc   = (float*)(ws + oBc);
    unsigned short* Wcb  = (unsigned short*)(ws + oWcb);
    unsigned short* Wevb = (unsigned short*)(ws + oWevb);
    unsigned short* vfB  = (unsigned short*)(ws + oVfB);
    float*          Whn  = (float*)(ws + oWhn);
    float*          fE   = (float*)(ws + oFe);
    unsigned short* fEB  = (unsigned short*)(ws + oFeB);
    float*          Whe  = (float*)(ws + oWhe);
    float*          outV = (float*)d_out;

    const long long w4  = (128 * 128) / 4;
    const long long nv4 = nvElems / 4;
    const long long ne4 = neElems / 4;

    // 1) fold layer-1 into the node->edge projection: Wc = Wve@W1, bc = Wve@b1 + bve
    fuse_weights<<<64, 256, 0, stream>>>(Wve, W1, Wc);
    fuse_bias<<<1, 128, 0, stream>>>(Wve, b1, bve, bc);

    // 2) precision conversions (bf16 operands for WMMA)
    cvt4_f32_bf16<<<(int)((w4  + 255) / 256), 256, 0, stream>>>(Wc,    Wcb,  w4);
    cvt4_f32_bf16<<<(int)((w4  + 255) / 256), 256, 0, stream>>>(Wev,   Wevb, w4);
    cvt4_f32_bf16<<<(int)((nv4 + 255) / 256), 256, 0, stream>>>(vfeat, vfB,  nv4);

    // 3) Wh_n = vfeat @ Wc^T + bc   (fused GEMM1+GEMM2)
    gemm_wmma_bf16<<<(NV + 127) / 128, 256, 0, stream>>>(vfB, Wcb, bc, Whn, NV);

    // 4) feat_e = segment_sum over hyperedges
    zero4_f32<<<(int)((ne4 + 255) / 256), 256, 0, stream>>>(fE, ne4);
    scatter_add<<<(NI + 7) / 8, 256, 0, stream>>>(Whn, node_idx, hedge_idx,
                                                  v_reg_weight, e_reg_sum, fE, NI);

    // 5) Wh_e = feat_e @ Wev^T + bev
    cvt4_f32_bf16<<<(int)((ne4 + 255) / 256), 256, 0, stream>>>(fE, fEB, ne4);
    gemm_wmma_bf16<<<(NE + 127) / 128, 256, 0, stream>>>(fEB, Wevb, bev, Whe, NE);

    // 6) feat_v_out = segment_sum over vertices
    zero4_f32<<<(int)((nv4 + 255) / 256), 256, 0, stream>>>(outV, nv4);
    scatter_add<<<(NI + 7) / 8, 256, 0, stream>>>(Whe, hedge_idx, node_idx,
                                                  e_reg_weight, v_reg_sum, outV, NI);
}
